// MultiWaveletTransform_25417616457773
// MI455X (gfx1250) — matrix operand order repaired and checked by reading, not compiled, not run on hardware
//
#include <hip/hip_runtime.h>
#include <math.h>

typedef __attribute__((ext_vector_type(16))) _Float16 v16h;
typedef __attribute__((ext_vector_type(8)))  _Float16 v8h;
typedef __attribute__((ext_vector_type(8)))  float    v8f;
typedef unsigned int u32x4 __attribute__((ext_vector_type(4)));
typedef int          i32x8 __attribute__((ext_vector_type(8)));
typedef int          i32x4 __attribute__((ext_vector_type(4)));

#define PI_F 3.14159265358979323846f

// ============================================================================
// Generic fp32 GEMM on the WMMA pipe with TDM (Tensor Data Mover) tile fetch.
//   C[M,N] = alpha * A[M,K] @ opB(B) + (accumulate ? C : 0) + bias[N]
// - Global->LDS tile movement is done by TENSOR_LOAD_TO_LDS (async DMA,
//   tracked on TENSORcnt), double buffered: descriptor for step s+1 is issued
//   before consuming step s, then s_wait_tensorcnt(2) + barrier.
// - The D# uses pad_enable so the DMA writes padded LDS rows (bank spread).
// - fp32 staged tiles are split into f16 hi+lo in LDS; each 16x16x32 product
//   uses 3 WMMAs (hi*hi + hi*lo + lo*hi) => ~fp32 accuracy at f16 rate.
// - Row remapping (r -> (r/rpg)*gst + r%rpg) on A is expressed directly in
//   the descriptor as a 3D tile (y=rpg rows, z=64/rpg groups, z-stride =
//   gst*lda); same remap on C handles the padded scatter-back.
// M,N multiples of 64; K multiple of 32.
// ============================================================================
#define BM 64
#define BN 64
#define BK 32
#define STR 48   // f16 LDS row stride in halfs: 96B (16B aligned, bank-spread)
#define SAS 36   // fp32 staging row stride for 32-wide tiles (TDM pad: 32+4)
#define SBS 68   // fp32 staging row stride for 64-wide tiles (TDM pad: 64+4)

// Build a D# (ISA 8.3/8.4) and issue TENSOR_LOAD_TO_LDS.
//   lds_off : LDS byte offset of destination
//   g       : global address of tile start
//   tw      : tile_dim0 (contiguous elements per row)
//   th      : tile_dim1 (rows, stride s0 elements)
//   tz      : tile_dim2 (row groups, stride s1 elements; 0 = unused/2D)
//   pic/pac : pad_interval / pad_amount codes (interval=2^(pic+1) DW, amt=pac+1 DW)
__device__ __forceinline__ void tdm_load_tile(unsigned lds_off, const float* g,
                                              unsigned tw, unsigned th, unsigned tz,
                                              unsigned long long s0,
                                              unsigned long long s1,
                                              unsigned pic, unsigned pac)
{
  unsigned long long ga = (unsigned long long)g;
  u32x4 g0;
  g0[0] = 1u;                                    // count=1, user descriptor
  g0[1] = lds_off;                               // lds_addr
  g0[2] = (unsigned)(ga & 0xffffffffu);          // global_addr[31:0]
  g0[3] = (unsigned)((ga >> 32) & 0x01ffffffu)   // global_addr[56:32]
        | (2u << 30);                            // type = 2 ("image")
  const unsigned td0 = 1u << 30, td1 = 1u << 30; // huge tensor dims: no OOB clip
  i32x8 g1;
  g1[0] = (int)((2u << 16)                       // data_size = 4 bytes
              | (1u << 20)                       // pad_enable
              | (pic << 22) | (pac << 25));
  g1[1] = (int)((td0 & 0xffffu) << 16);          // tensor_dim0[15:0]  (atomic addr = 0)
  g1[2] = (int)(((td0 >> 16) & 0xffffu) | ((td1 & 0xffffu) << 16));
  g1[3] = (int)(((td1 >> 16) & 0xffffu) | (tw << 16));   // tile_dim0
  g1[4] = (int)((th & 0xffffu) | (tz << 16));            // tile_dim1, tile_dim2
  g1[5] = (int)(s0 & 0xffffffffu);                       // tensor_dim0_stride lo
  g1[6] = (int)(((s0 >> 32) & 0xffffu) | ((s1 & 0xffffu) << 16));
  g1[7] = (int)((s1 >> 16) & 0xffffffffu);               // tensor_dim1_stride hi
  i32x4 z4 = {0, 0, 0, 0};
  i32x8 z8 = {0, 0, 0, 0, 0, 0, 0, 0};
  __builtin_amdgcn_tensor_load_to_lds(g0, g1, z4, z4, z8, 0);
}

__global__ __launch_bounds__(128)
void gemm_f32_wmma(const float* __restrict__ A, const float* __restrict__ B,
                   float* __restrict__ C, const float* __restrict__ bias,
                   int M, int N, int K, int lda, int ldb, int ldc,
                   int opBT, float alpha, int accumulate,
                   int a_rpg, int a_gst, int c_rpg, int c_gst)
{
  __shared__ float Sa[2][BM * SAS];              // TDM-padded fp32 A tiles
  __shared__ float Sb[2][BM * SAS];              // fp32 B tiles (both layouts fit)
  __shared__ _Float16 Ah[BM * STR], Al[BM * STR];
  __shared__ _Float16 Bh[BN * STR], Bl[BN * STR];

  const int tid  = threadIdx.x;
  const int wave = tid >> 5;
  const int lane = tid & 31;
  const int l16  = lane & 15;
  const int hv   = lane >> 4;
  const int bm   = blockIdx.y * BM;
  const int bn   = blockIdx.x * BN;
  const int steps = K / BK;

  // A-tile shape under the row remap (uniform across the block).
  unsigned th_a, tz_a;
  unsigned long long s1_a;
  if (a_rpg >= BM) { th_a = BM;             tz_a = 0;                     s1_a = 0; }
  else             { th_a = (unsigned)a_rpg; tz_a = (unsigned)(BM/a_rpg); s1_a = (unsigned long long)a_gst * (unsigned)lda; }
  const int prow0 = (bm / a_rpg) * a_gst + (bm % a_rpg);

  const unsigned lds_a0 = (unsigned)(unsigned long long)&Sa[0][0];
  const unsigned lds_a1 = (unsigned)(unsigned long long)&Sa[1][0];
  const unsigned lds_b0 = (unsigned)(unsigned long long)&Sb[0][0];
  const unsigned lds_b1 = (unsigned)(unsigned long long)&Sb[1][0];

  v8f acc[4] = {};

  // Prologue: DMA tiles for step 0 (wave 0 owns the TDM / TENSORcnt).
  if (wave == 0) {
    const float* ga = A + (unsigned long long)prow0 * lda + 0;
    tdm_load_tile(lds_a0, ga, BK, th_a, tz_a, (unsigned)lda, s1_a, 4u, 3u);
    if (opBT) tdm_load_tile(lds_b0, B + (unsigned long long)bn * ldb + 0, BK, BN, 0,
                            (unsigned)ldb, 0, 4u, 3u);
    else      tdm_load_tile(lds_b0, B + 0 * (unsigned long long)ldb + bn, BN, BK, 0,
                            (unsigned)ldb, 0, 5u, 3u);
  }

  for (int s = 0; s < steps; ++s) {
    const int buf = s & 1;
    if (wave == 0) {
      if (s + 1 < steps) {
        // Issue next tiles, then wait for the *oldest* pair only (<=2 newer).
        const int k1 = (s + 1) * BK;
        const unsigned la = (buf ? lds_a0 : lds_a1), lb = (buf ? lds_b0 : lds_b1);
        const float* ga = A + (unsigned long long)prow0 * lda + k1;
        tdm_load_tile(la, ga, BK, th_a, tz_a, (unsigned)lda, s1_a, 4u, 3u);
        if (opBT) tdm_load_tile(lb, B + (unsigned long long)bn * ldb + k1, BK, BN, 0,
                                (unsigned)ldb, 0, 4u, 3u);
        else      tdm_load_tile(lb, B + (unsigned long long)k1 * ldb + bn, BN, BK, 0,
                                (unsigned)ldb, 0, 5u, 3u);
        __builtin_amdgcn_s_wait_tensorcnt(2);
      } else {
        __builtin_amdgcn_s_wait_tensorcnt(0);
      }
    }
    __syncthreads();   // fp32 tiles for step s are in LDS for all waves

    // ---- split fp32 staging -> f16 hi/lo fragment layout ----
    {
      int i   = tid >> 1;
      int seg = (tid & 1) * 16;
      const float* src = &Sa[buf][i * SAS + seg];
      _Float16* dh = Ah + i * STR + seg;
      _Float16* dl = Al + i * STR + seg;
#pragma unroll
      for (int q = 0; q < 4; ++q) {
        float4 v = *(const float4*)(src + q * 4);
        float e0 = v.x, e1 = v.y, e2 = v.z, e3 = v.w;
        _Float16 h0 = (_Float16)e0, h1 = (_Float16)e1;
        _Float16 h2 = (_Float16)e2, h3 = (_Float16)e3;
        dh[q*4+0] = h0; dh[q*4+1] = h1; dh[q*4+2] = h2; dh[q*4+3] = h3;
        dl[q*4+0] = (_Float16)(e0 - (float)h0);
        dl[q*4+1] = (_Float16)(e1 - (float)h1);
        dl[q*4+2] = (_Float16)(e2 - (float)h2);
        dl[q*4+3] = (_Float16)(e3 - (float)h3);
      }
    }
    if (opBT) {          // staged [N x K] rows: direct copy to N-major f16
      int j   = tid >> 1;
      int seg = (tid & 1) * 16;
      const float* src = &Sb[buf][j * SAS + seg];
      _Float16* dh = Bh + j * STR + seg;
      _Float16* dl = Bl + j * STR + seg;
#pragma unroll
      for (int q = 0; q < 4; ++q) {
        float4 v = *(const float4*)(src + q * 4);
        float e0 = v.x, e1 = v.y, e2 = v.z, e3 = v.w;
        _Float16 h0 = (_Float16)e0, h1 = (_Float16)e1;
        _Float16 h2 = (_Float16)e2, h3 = (_Float16)e3;
        dh[q*4+0] = h0; dh[q*4+1] = h1; dh[q*4+2] = h2; dh[q*4+3] = h3;
        dl[q*4+0] = (_Float16)(e0 - (float)h0);
        dl[q*4+1] = (_Float16)(e1 - (float)h1);
        dl[q*4+2] = (_Float16)(e2 - (float)h2);
        dl[q*4+3] = (_Float16)(e3 - (float)h3);
      }
    } else {             // staged [K x N] rows: transpose while splitting
      int kk = tid >> 2;
      int j0 = (tid & 3) * 16;
      const float* src = &Sb[buf][kk * SBS + j0];
#pragma unroll
      for (int q = 0; q < 4; ++q) {
        float4 v = *(const float4*)(src + q * 4);
        float es[4] = {v.x, v.y, v.z, v.w};
#pragma unroll
        for (int e = 0; e < 4; ++e) {
          int j = j0 + q * 4 + e;
          _Float16 h = (_Float16)es[e];
          Bh[j * STR + kk] = h;
          Bl[j * STR + kk] = (_Float16)(es[e] - (float)h);
        }
      }
    }
    __syncthreads();

    // ---- fragments (ISA 7.12.2 layouts) + 12 WMMAs ----
    const v8h* pah = (const v8h*)(Ah + (wave * 16 + l16) * STR + hv * 8);
    const v8h* pal = (const v8h*)(Al + (wave * 16 + l16) * STR + hv * 8);
    v8h ah0 = pah[0], ah1 = pah[2];
    v8h al0 = pal[0], al1 = pal[2];
    v16h a_hi = __builtin_shufflevector(ah0, ah1, 0,1,2,3,4,5,6,7,8,9,10,11,12,13,14,15);
    v16h a_lo = __builtin_shufflevector(al0, al1, 0,1,2,3,4,5,6,7,8,9,10,11,12,13,14,15);

#pragma unroll
    for (int t = 0; t < 4; ++t) {
      const v8h* pbh = (const v8h*)(Bh + (t * 16 + l16) * STR + hv * 8);
      const v8h* pbl = (const v8h*)(Bl + (t * 16 + l16) * STR + hv * 8);
      v8h bh0 = pbh[0], bh1 = pbh[2];
      v8h bl0 = pbl[0], bl1 = pbl[2];
      v16h b_hi = __builtin_shufflevector(bh0, bh1, 0,1,2,3,4,5,6,7,8,9,10,11,12,13,14,15);
      v16h b_lo = __builtin_shufflevector(bl0, bl1, 0,1,2,3,4,5,6,7,8,9,10,11,12,13,14,15);

      acc[t] = __builtin_amdgcn_wmma_f32_16x16x32_f16(false, a_hi, false, b_hi,
                                                      (short)0, acc[t], false, false);
      acc[t] = __builtin_amdgcn_wmma_f32_16x16x32_f16(false, a_hi, false, b_lo,
                                                      (short)0, acc[t], false, false);
      acc[t] = __builtin_amdgcn_wmma_f32_16x16x32_f16(false, a_lo, false, b_hi,
                                                      (short)0, acc[t], false, false);
    }
    __syncthreads();     // protect f16 arrays before next step's convert
  }

  // ---- epilogue: C frag: lane<16 -> N=l16, M=r; lane>=16 -> M=8+r ----
#pragma unroll
  for (int t = 0; t < 4; ++t) {
    int n = bn + t * 16 + l16;
    float bv = bias ? bias[n] : 0.0f;
#pragma unroll
    for (int r = 0; r < 8; ++r) {
      int m  = bm + wave * 16 + hv * 8 + r;
      int pm = (m / c_rpg) * c_gst + (m % c_rpg);
      float* cp = C + (size_t)pm * ldc + n;
      float v = alpha * acc[t][r] + bv;
      if (accumulate) v += *cp;
      *cp = v;
    }
  }
}

// ============================================================================
// Elementwise plumbing for DCT-II / IDCT-II of length 32768 via four-step FFT
// (N = N1*N2 = 128*256), Makhoul even/odd mapping.  All O(N) passes.
// ============================================================================
__global__ void k_zero(float* p, int n) {
  int i = blockIdx.x * blockDim.x + threadIdx.x;
  if (i < n) p[i] = 0.0f;
}

__global__ void k_build_dct(float* D, int L) {
  int i = blockIdx.x * blockDim.x + threadIdx.x;
  if (i >= L * L) return;
  int k = i / L, n = i % L;
  int t = ((2 * n + 1) * k) % (4 * L);
  float s = (k == 0) ? sqrtf(1.0f / L) : sqrtf(2.0f / L);
  D[i] = s * cosf(PI_F * (float)t / (2.0f * L));
}

__global__ void k_build_fft(float* Fr, float* Fi, int n) {
  int i = blockIdx.x * blockDim.x + threadIdx.x;
  if (i >= n * n) return;
  int k = i / n, j = i % n;
  int m = (k * j) % n;
  float th = 2.0f * PI_F * (float)m / (float)n;
  Fr[i] = cosf(th);
  Fi[i] = -sinf(th);
}

__global__ void k_reorder_fwd(const float* __restrict__ X, float* __restrict__ Pr,
                              int cols, int Nf) {
  int i = blockIdx.x * blockDim.x + threadIdx.x;
  if (i >= Nf * cols) return;
  int n = i / cols, c = i % cols;
  int src = (n < Nf / 2) ? (2 * n) : (2 * (Nf - 1 - n) + 1);
  Pr[i] = X[(size_t)src * cols + c];
}

__global__ void k_twiddle(float* Pr, float* Pi, int cols) {
  int i = blockIdx.x * blockDim.x + threadIdx.x;
  int tot = 128 * 256 * cols;
  if (i >= tot) return;
  int k1 = i / (256 * cols);
  int n2 = (i / cols) % 256;
  float th = -2.0f * PI_F * (float)(k1 * n2) / 32768.0f;
  float cth = cosf(th), sth = sinf(th);
  float ar = Pr[i], ai = Pi[i];
  Pr[i] = ar * cth - ai * sth;
  Pi[i] = ar * sth + ai * cth;
}

__global__ void k_transpose(const float* __restrict__ Sr, const float* __restrict__ Si,
                            float* __restrict__ Dr, float* __restrict__ Di, int cols) {
  int i = blockIdx.x * blockDim.x + threadIdx.x;
  int tot = 128 * 256 * cols;
  if (i >= tot) return;
  int n2 = i / (128 * cols);
  int k1 = (i / cols) % 128;
  int c  = i % cols;
  size_t s = (size_t)k1 * 256 * cols + (size_t)n2 * cols + c;
  Dr[i] = Sr[s];
  Di[i] = Si[s];
}

__global__ void k_post_dct(const float* __restrict__ Pr, const float* __restrict__ Pi,
                           float* __restrict__ T, int cols, int Nf) {
  int i = blockIdx.x * blockDim.x + threadIdx.x;
  if (i >= Nf * cols) return;
  int k = i / cols, c = i % cols;
  int k1 = k & 127, k2 = k >> 7;
  size_t s = (size_t)k2 * 128 * cols + (size_t)k1 * cols + c;
  float phi = 0.5f * PI_F * (float)k / (float)Nf;
  float sc  = (k == 0) ? rsqrtf((float)Nf) : sqrtf(2.0f / (float)Nf);
  T[i] = sc * (cosf(phi) * Pr[s] + sinf(phi) * Pi[s]);
}

__global__ void k_pre_idct(const float* __restrict__ Y, float* __restrict__ Pr,
                           float* __restrict__ Pi, int cols, int Nf) {
  int i = blockIdx.x * blockDim.x + threadIdx.x;
  if (i >= Nf * cols) return;
  int k = i / cols, c = i % cols;
  float sc = (k == 0) ? rsqrtf((float)Nf) : sqrtf(2.0f / (float)Nf);
  float u = Y[i] / sc;
  float w = (k == 0) ? 0.0f : (Y[(size_t)(Nf - k) * cols + c] / sc);
  float phi = 0.5f * PI_F * (float)k / (float)Nf;
  float cp = cosf(phi), sp = sinf(phi);
  Pr[i] =  (u * cp + w * sp);
  Pi[i] = -(u * sp - w * cp);     // conjugate for IFFT-via-FFT
}

__global__ void k_post_idct(const float* __restrict__ Pr, float* __restrict__ out,
                            int cols, int Nf) {
  int i = blockIdx.x * blockDim.x + threadIdx.x;
  if (i >= Nf * cols) return;
  int m = i / cols, c = i % cols;
  int k1 = m & 127, k2 = m >> 7;
  size_t s = (size_t)k2 * 128 * cols + (size_t)k1 * cols + c;
  float vr = Pr[s] / (float)Nf;
  int row = (m < Nf / 2) ? (2 * m) : (2 * (Nf - 1 - m) + 1);
  out[(size_t)row * cols + c] = vr;
}

// ============================================================================
// Host orchestration.  Workspace ~1.15 GB of fp32 (trivial vs 432 GiB HBM).
// ============================================================================
extern "C" void kernel_launch(void* const* d_in, const int* in_sizes, int n_in,
                              void* d_out, int out_size, void* d_ws, size_t ws_size,
                              hipStream_t stream)
{
  (void)in_sizes; (void)n_in; (void)out_size; (void)ws_size;
  const float* x  = (const float*)d_in[0];
  const float* W1 = (const float*)d_in[1];
  const float* b1 = (const float*)d_in[2];
  const float* W2 = (const float*)d_in[3];
  const float* b2 = (const float*)d_in[4];
  const float* W3 = (const float*)d_in[5];
  const float* b3 = (const float*)d_in[6];
  float* out = (float*)d_out;

  const int Bb = 512, Cc = 64, L = 512, NN = 32768;
  const int KL = 256, KC = 32, KL2 = 128, KC2 = 16;
  const int N1 = 128, N2 = 256;
  const int CF = 512;
  const int CI = 3 * 512;

  float* ws = (float*)d_ws;
  size_t off = 0;
  auto alloc = [&](size_t n) { float* p = ws + off; off += (n + 63) & ~(size_t)63; return p; };
  float* Dm  = alloc((size_t)L * L);
  float* F1r = alloc((size_t)N1 * N1);
  float* F1i = alloc((size_t)N1 * N1);
  float* F2r = alloc((size_t)N2 * N2);
  float* F2i = alloc((size_t)N2 * N2);
  float* Tb  = alloc((size_t)NN * L);
  float* Ub  = alloc((size_t)NN * L);
  float* Yc  = alloc((size_t)NN * CI);
  const size_t PL = (size_t)NN * CI;
  float* P0r = alloc(PL); float* P0i = alloc(PL);
  float* P1r = alloc(PL); float* P1i = alloc(PL);

  auto blocks = [](int n) { return dim3((unsigned)((n + 255) / 256)); };

  auto gemm = [&](const float* A, const float* Bm, float* Cm, const float* bias,
                  int M, int N, int K, int lda, int ldb, int ldc, int opBT,
                  float alpha, int accum, int arpg, int agst, int crpg, int cgst) {
    dim3 g(N / 64, M / 64), blk(128);
    gemm_f32_wmma<<<g, blk, 0, stream>>>(A, Bm, Cm, bias, M, N, K, lda, ldb, ldc,
                                         opBT, alpha, accum, arpg, agst, crpg, cgst);
  };

  // ---- operator bases ----
  k_build_dct<<<blocks(L * L), 256, 0, stream>>>(Dm, L);
  k_build_fft<<<blocks(N1 * N1), 256, 0, stream>>>(F1r, F1i, N1);
  k_build_fft<<<blocks(N2 * N2), 256, 0, stream>>>(F2r, F2i, N2);

  // ---- axis-1 DCT: Ub = X @ D^T ----
  gemm(x, Dm, Ub, nullptr, NN, L, L, L, L, L, 1, 1.f, 0, NN, 0, NN, 0);

  // ---- axis-0 DCT via four-step FFT (cols = 512) ----
  k_reorder_fwd<<<blocks(NN * CF), 256, 0, stream>>>(Ub, P0r, CF, NN);
  gemm(F1r, P0r, P1r, nullptr, N1, N2 * CF, N1, N1, N2 * CF, N2 * CF, 0, 1.f, 0, N1, 0, N1, 0);
  gemm(F1i, P0r, P1i, nullptr, N1, N2 * CF, N1, N1, N2 * CF, N2 * CF, 0, 1.f, 0, N1, 0, N1, 0);
  k_twiddle<<<blocks(N1 * N2 * CF), 256, 0, stream>>>(P1r, P1i, CF);
  k_transpose<<<blocks(N1 * N2 * CF), 256, 0, stream>>>(P1r, P1i, P0r, P0i, CF);
  gemm(F2r, P0r, P1r, nullptr, N2, N1 * CF, N2, N2, N1 * CF, N1 * CF, 0,  1.f, 0, N2, 0, N2, 0);
  gemm(F2i, P0i, P1r, nullptr, N2, N1 * CF, N2, N2, N1 * CF, N1 * CF, 0, -1.f, 1, N2, 0, N2, 0);
  gemm(F2r, P0i, P1i, nullptr, N2, N1 * CF, N2, N2, N1 * CF, N1 * CF, 0,  1.f, 0, N2, 0, N2, 0);
  gemm(F2i, P0r, P1i, nullptr, N2, N1 * CF, N2, N2, N1 * CF, N1 * CF, 0,  1.f, 1, N2, 0, N2, 0);
  k_post_dct<<<blocks(NN * CF), 256, 0, stream>>>(P1r, P1i, Tb, CF, NN);

  // ---- spectral linear layers + axis-1 IDCTs into concat buffer ----
  k_zero<<<blocks(NN * CI), 256, 0, stream>>>(Yc, NN * CI);
  gemm(Tb, W1, Ub, b1, NN, L, L, L, L, L, 1, 1.f, 0, NN, 0, NN, 0);
  gemm(Ub, Dm, Yc + 0, nullptr, NN, L, L, L, L, CI, 0, 1.f, 0, NN, 0, NN, 0);
  gemm(Tb, W2, Ub, b2, Bb * KC, KL, KL, L, KL, KL, 1, 1.f, 0, KC, Cc, Bb * KC, 0);
  gemm(Ub, Dm, Yc + L, nullptr, Bb * KC, L, KL, KL, L, CI, 0, 1.f, 0, Bb * KC, 0, KC, Cc);
  gemm(Tb, W3, Ub, b3, Bb * KC2, KL2, KL2, L, KL2, KL2, 1, 1.f, 0, KC2, Cc, Bb * KC2, 0);
  gemm(Ub, Dm, Yc + 2 * L, nullptr, Bb * KC2, L, KL2, KL2, L, CI, 0, 1.f, 0, Bb * KC2, 0, KC2, Cc);

  // ---- axis-0 IDCT on the 1536-wide concat via the same FFT machinery ----
  k_pre_idct<<<blocks(NN * CI), 256, 0, stream>>>(Yc, P0r, P0i, CI, NN);
  gemm(F1r, P0r, P1r, nullptr, N1, N2 * CI, N1, N1, N2 * CI, N2 * CI, 0,  1.f, 0, N1, 0, N1, 0);
  gemm(F1i, P0i, P1r, nullptr, N1, N2 * CI, N1, N1, N2 * CI, N2 * CI, 0, -1.f, 1, N1, 0, N1, 0);
  gemm(F1r, P0i, P1i, nullptr, N1, N2 * CI, N1, N1, N2 * CI, N2 * CI, 0,  1.f, 0, N1, 0, N1, 0);
  gemm(F1i, P0r, P1i, nullptr, N1, N2 * CI, N1, N1, N2 * CI, N2 * CI, 0,  1.f, 1, N1, 0, N1, 0);
  k_twiddle<<<blocks(N1 * N2 * CI), 256, 0, stream>>>(P1r, P1i, CI);
  k_transpose<<<blocks(N1 * N2 * CI), 256, 0, stream>>>(P1r, P1i, P0r, P0i, CI);
  gemm(F2r, P0r, P1r, nullptr, N2, N1 * CI, N2, N2, N1 * CI, N1 * CI, 0,  1.f, 0, N2, 0, N2, 0);
  gemm(F2i, P0i, P1r, nullptr, N2, N1 * CI, N2, N2, N1 * CI, N1 * CI, 0, -1.f, 1, N2, 0, N2, 0);
  gemm(F2r, P0i, P1i, nullptr, N2, N1 * CI, N2, N2, N1 * CI, N1 * CI, 0,  1.f, 0, N2, 0, N2, 0);
  gemm(F2i, P0r, P1i, nullptr, N2, N1 * CI, N2, N2, N1 * CI, N1 * CI, 0,  1.f, 1, N2, 0, N2, 0);
  k_post_idct<<<blocks(NN * CI), 256, 0, stream>>>(P1r, out, CI, NN);
}
